// GraphAttentionLayer_89489938579705
// MI455X (gfx1250) — compile-verified, matrix operand order checked by reading, not run on hardware
//
#include <hip/hip_runtime.h>
#include <hip/hip_bf16.h>

#define ALPHA 0.2f
#define MASK_VAL -1000000000.0f

typedef __bf16 v16bf __attribute__((ext_vector_type(16)));
typedef __bf16 v8bf  __attribute__((ext_vector_type(8)));
typedef float  v8f   __attribute__((ext_vector_type(8)));
typedef unsigned int u32x4 __attribute__((ext_vector_type(4)));
typedef int          i32x4 __attribute__((ext_vector_type(4)));
typedef int          i32x8 __attribute__((ext_vector_type(8)));

union F8      { v8f   v; float  e[8];  };
union BF16x16 { v16bf v; v8bf h[2]; __bf16 e[16]; };

#if __has_builtin(__builtin_amdgcn_tensor_load_to_lds) && \
    __has_builtin(__builtin_amdgcn_s_wait_tensorcnt)
#define USE_TDM 1
#else
#define USE_TDM 0
#endif

// ---------------------------------------------------------------------------
// K1: Wh = h @ W   (M=16384, N=512, K=512).
// Block tile 64x64, BK=32, 8 waves, wave tile 16x32 (2 WMMA C-tiles).
// LDS operands stored bf16, K-contiguous (stride 40) -> fragments are
// 2x ds_load_b128 each.  Outputs Wh row-major AND per-batch transposed.
// ---------------------------------------------------------------------------
__global__ __launch_bounds__(256) void k1_gemm(const float* __restrict__ h,
                                               const float* __restrict__ W,
                                               __bf16* __restrict__ Whb,
                                               __bf16* __restrict__ Whbt) {
    __shared__ __bf16 hs[64 * 40];   // A: [m][k]  stride 40 (80B rows)
    __shared__ __bf16 wt[64 * 40];   // B: [n][k]  stride 40 (transposed W)
    __shared__ __bf16 cs[64 * 72];   // C staging for transposed store

    const int t    = threadIdx.x;
    const int w    = t >> 5;
    const int lane = t & 31;
    const int lr   = lane & 15;
    const int lh   = lane >> 4;
    const int m0   = blockIdx.x * 64;
    const int n0   = blockIdx.y * 64;
    const int wm   = w & 3;   // wave M tile (16 rows)
    const int wn   = w >> 2;  // wave N tile (32 cols)

    F8 acc[2];
#pragma unroll
    for (int nt = 0; nt < 2; nt++)
#pragma unroll
        for (int r = 0; r < 8; r++) acc[nt].e[r] = 0.f;

    for (int k0 = 0; k0 < 512; k0 += 32) {
        { // stage h tile 64x32 -> bf16 [m][k]
            const int hr = t >> 2;
            const int c  = (t & 3) * 8;
            const float4* src = reinterpret_cast<const float4*>(
                h + (size_t)(m0 + hr) * 512 + k0 + c);
            const float4 x0 = src[0], x1 = src[1];
            __bf16* d = &hs[hr * 40 + c];
            d[0] = (__bf16)x0.x; d[1] = (__bf16)x0.y;
            d[2] = (__bf16)x0.z; d[3] = (__bf16)x0.w;
            d[4] = (__bf16)x1.x; d[5] = (__bf16)x1.y;
            d[6] = (__bf16)x1.z; d[7] = (__bf16)x1.w;
        }
        { // stage W tile 32x64 -> bf16 transposed [n][k]
            const int kr = t >> 3;
            const int c  = (t & 7) * 8;
            const float4* src = reinterpret_cast<const float4*>(
                W + (size_t)(k0 + kr) * 512 + n0 + c);
            const float4 x0 = src[0], x1 = src[1];
            wt[(c + 0) * 40 + kr] = (__bf16)x0.x;
            wt[(c + 1) * 40 + kr] = (__bf16)x0.y;
            wt[(c + 2) * 40 + kr] = (__bf16)x0.z;
            wt[(c + 3) * 40 + kr] = (__bf16)x0.w;
            wt[(c + 4) * 40 + kr] = (__bf16)x1.x;
            wt[(c + 5) * 40 + kr] = (__bf16)x1.y;
            wt[(c + 6) * 40 + kr] = (__bf16)x1.z;
            wt[(c + 7) * 40 + kr] = (__bf16)x1.w;
        }
        __syncthreads();

        BF16x16 af;
        const int mm = 16 * wm + lr;
        af.h[0] = *reinterpret_cast<const v8bf*>(&hs[mm * 40 + 8 * lh]);
        af.h[1] = *reinterpret_cast<const v8bf*>(&hs[mm * 40 + 16 + 8 * lh]);

#pragma unroll
        for (int nt = 0; nt < 2; nt++) {
            BF16x16 bf;
            const int nn = 32 * wn + 16 * nt + lr;
            bf.h[0] = *reinterpret_cast<const v8bf*>(&wt[nn * 40 + 8 * lh]);
            bf.h[1] = *reinterpret_cast<const v8bf*>(&wt[nn * 40 + 16 + 8 * lh]);
            acc[nt].v = __builtin_amdgcn_wmma_f32_16x16x32_bf16(
                false, af.v, false, bf.v, (short)0, acc[nt].v, false, false);
        }
        __syncthreads();
    }

    // row-major store (consumed by k2) + LDS transpose staging
#pragma unroll
    for (int nt = 0; nt < 2; nt++)
#pragma unroll
        for (int r = 0; r < 8; r++) {
            const int m_l = 16 * wm + r + 8 * lh;
            const int n_l = 32 * wn + 16 * nt + lr;
            const __bf16 x = (__bf16)acc[nt].e[r];
            Whb[(size_t)(m0 + m_l) * 512 + n0 + n_l] = x;
            cs[n_l * 72 + m_l] = x;
        }
    __syncthreads();

    { // coalesced store of transposed tile: Whbt[b][n][i]
        const int b1     = m0 >> 11;
        const int i_base = (m0 & 2047) + (t & 3) * 16;
        const int n_l    = t >> 2;
        const uint4* src = reinterpret_cast<const uint4*>(&cs[n_l * 72 + (t & 3) * 16]);
        uint4* dst = reinterpret_cast<uint4*>(
            Whbt + ((size_t)b1 * 512 + n0 + n_l) * 2048 + i_base);
        dst[0] = src[0];
        dst[1] = src[1];
    }
}

// ---------------------------------------------------------------------------
// K2: e_i = Wh . a1, e_j = Wh . a2  (one wave per row, row-major Whb)
// ---------------------------------------------------------------------------
__global__ __launch_bounds__(256) void k2_rowdots(const __bf16* __restrict__ Whb,
                                                  const float* __restrict__ a,
                                                  float* __restrict__ ei,
                                                  float* __restrict__ ej) {
    const int w    = threadIdx.x >> 5;
    const int lane = threadIdx.x & 31;
    const int row  = blockIdx.x * 8 + w;
    const __bf16* src = Whb + (size_t)row * 512;
    float s1 = 0.f, s2 = 0.f;
#pragma unroll
    for (int i = 0; i < 16; i++) {
        const int c = lane + 32 * i;
        const float x = (float)src[c];
        s1 += x * a[c];
        s2 += x * a[512 + c];
    }
#pragma unroll
    for (int off = 16; off; off >>= 1) {
        s1 += __shfl_xor(s1, off, 32);
        s2 += __shfl_xor(s2, off, 32);
    }
    if (lane == 0) { ei[row] = s1; ej[row] = s2; }
}

// ---------------------------------------------------------------------------
// K3: per-row softmax stats (max m, 1/sum).  One wave per row, int4 adj reads.
// ---------------------------------------------------------------------------
__global__ __launch_bounds__(256) void k3_stats(const int* __restrict__ adj,
                                                const float* __restrict__ ei,
                                                const float* __restrict__ ej,
                                                float* __restrict__ rmax,
                                                float* __restrict__ rsinv) {
    const int w    = threadIdx.x >> 5;
    const int lane = threadIdx.x & 31;
    const int row  = blockIdx.x * 8 + w;       // 0..16383
    const int b    = row >> 11;
    const int i    = row & 2047;
    const int* arow = adj + (size_t)b * 2048 * 2048 + (size_t)i * 2048;
    const float* ejb = ej + ((size_t)b << 11);
    const float eiv = ei[row];

    float m = -3.0e38f, s = 0.f;
    for (int j = lane * 4; j < 2048; j += 128) {
        const int4   av  = *reinterpret_cast<const int4*>(arow + j);
        const float4 ejv = *reinterpret_cast<const float4*>(ejb + j);
        float e[4] = {eiv + ejv.x, eiv + ejv.y, eiv + ejv.z, eiv + ejv.w};
        const int msk[4] = {av.x, av.y, av.z, av.w};
#pragma unroll
        for (int q = 0; q < 4; q++) {
            float ev = (e[q] > 0.f) ? e[q] : ALPHA * e[q];
            if (msk[q] == 0) ev = MASK_VAL;
            const float mn = fmaxf(m, ev);
            s = s * __expf(m - mn) + __expf(ev - mn);
            m = mn;
        }
    }
#pragma unroll
    for (int off = 16; off; off >>= 1) {
        const float mo = __shfl_xor(m, off, 32);
        const float so = __shfl_xor(s, off, 32);
        const float mn = fmaxf(m, mo);
        s = s * __expf(m - mn) + so * __expf(mo - mn);
        m = mn;
    }
    if (lane == 0) { rmax[row] = m; rsinv[row] = 1.f / s; }
}

// ---------------------------------------------------------------------------
// K4: out = elu( softmax(e) @ Wh ).  WG owns 32 rows; j streamed in 32-chunks.
// Wh chunk (512 cols x 32 k) DMA'd by the Tensor Data Mover from the
// transposed copy, double-buffered, with TDM padding to an 80B LDS row
// stride so every WMMA fragment is 2x ds_load_b128.
// ---------------------------------------------------------------------------
#define WHS_ROW 40                      // bf16 per LDS row (64B data + 16B pad)

#if USE_TDM
__device__ __forceinline__ void tdm_issue(unsigned lds_off,
                                          const __bf16* gsrc) {
    const unsigned long long ga = (unsigned long long)(uintptr_t)gsrc;
    const u32x4 g0 = {1u,                                   // count=1
                      lds_off,                              // lds_addr
                      (unsigned)ga,                         // global_addr lo
                      (unsigned)((ga >> 32) & 0x01FFFFFFull) | (2u << 30)};
    // data_size=1(2B), pad_enable, pad_interval=16DW, pad_amount=4DW
    const i32x8 g1 = {(int)((1u << 16) | (1u << 20) | (3u << 22) | (3u << 25)),
                      (int)(2048u << 16),   // tensor_dim0 = 2048 (lo16<<16)
                      (int)(512u << 16),    // td0 hi | tensor_dim1 = 512
                      (int)(32u << 16),     // td1 hi | tile_dim0 = 32
                      512,                  // tile_dim1 = 512, tile_dim2 = 0
                      2048,                 // tensor_dim0_stride lo32
                      0, 0};
    const i32x4 gz = {0, 0, 0, 0};
#if __clang_major__ >= 23
    const i32x8 gz8 = {0, 0, 0, 0, 0, 0, 0, 0};
    __builtin_amdgcn_tensor_load_to_lds(g0, g1, gz, gz, gz8, 0);
#else
    __builtin_amdgcn_tensor_load_to_lds(g0, g1, gz, gz, 0);
#endif
}
#endif

__global__ __launch_bounds__(256) void k4_attn(const int* __restrict__ adj,
                                               const __bf16* __restrict__ Whbt,
                                               const float* __restrict__ ei,
                                               const float* __restrict__ ej,
                                               const float* __restrict__ rmax,
                                               const float* __restrict__ rsinv,
                                               float* __restrict__ out) {
    __shared__ __bf16 whs[2][512 * WHS_ROW];  // Wh chunk [n][k], 80B rows
    __shared__ __bf16 pblk[32 * WHS_ROW];     // P block  [i][j], bf16

    const int t    = threadIdx.x;
    const int w    = t >> 5;
    const int lane = t & 31;
    const int lr   = lane & 15;
    const int lh   = lane >> 4;
    const int row0 = blockIdx.x * 32;     // flat row (b*2048 + i0)
    const int b    = row0 >> 11;
    const int n0   = w * 64;              // wave's output column base

    // per-thread P-row constants
    const int pr  = t >> 3;               // row in tile
    const int pc0 = (t & 7) * 4;          // col in j-chunk
    const int rg  = row0 + pr;
    const float eiv = ei[rg];
    const float mv  = rmax[rg];
    const float isv = rsinv[rg];
    const int* arow = adj + (size_t)b * 2048 * 2048 + (size_t)(rg & 2047) * 2048;
    const float* ejb = ej + ((size_t)b << 11);
    const __bf16* gbase = Whbt + (size_t)b * 512 * 2048;

    F8 acc[2][4];
#pragma unroll
    for (int mt = 0; mt < 2; mt++)
#pragma unroll
        for (int nt = 0; nt < 4; nt++)
#pragma unroll
            for (int r = 0; r < 8; r++) acc[mt][nt].e[r] = 0.f;

#if USE_TDM
    if (w == 0) tdm_issue((unsigned)(uintptr_t)&whs[0][0], gbase);
#endif

    for (int j0 = 0; j0 < 2048; j0 += 32) {
        const int cur = (j0 >> 5) & 1;
#if USE_TDM
        if (w == 0) {
            if (j0 + 32 < 2048) {
                tdm_issue((unsigned)(uintptr_t)&whs[cur ^ 1][0], gbase + (j0 + 32));
                __builtin_amdgcn_s_wait_tensorcnt(1);
            } else {
                __builtin_amdgcn_s_wait_tensorcnt(0);
            }
        }
#else
        { // manual staging fallback: 2 rows of 32 bf16 per thread
            const int n = t * 2;
#pragma unroll
            for (int q = 0; q < 2; q++) {
                const uint4* src = reinterpret_cast<const uint4*>(
                    gbase + (size_t)(n + q) * 2048 + j0);
                uint4* dst = reinterpret_cast<uint4*>(&whs[cur][(n + q) * WHS_ROW]);
                dst[0] = src[0]; dst[1] = src[1];
                dst[2] = src[2]; dst[3] = src[3];
            }
        }
#endif
        { // normalized attention probabilities, stored bf16 K-contiguous
            const int4   av  = *reinterpret_cast<const int4*>(arow + j0 + pc0);
            const float4 ejv = *reinterpret_cast<const float4*>(ejb + j0 + pc0);
            float e0 = eiv + ejv.x, e1 = eiv + ejv.y, e2 = eiv + ejv.z, e3 = eiv + ejv.w;
            e0 = (e0 > 0.f) ? e0 : ALPHA * e0; if (av.x == 0) e0 = MASK_VAL;
            e1 = (e1 > 0.f) ? e1 : ALPHA * e1; if (av.y == 0) e1 = MASK_VAL;
            e2 = (e2 > 0.f) ? e2 : ALPHA * e2; if (av.z == 0) e2 = MASK_VAL;
            e3 = (e3 > 0.f) ? e3 : ALPHA * e3; if (av.w == 0) e3 = MASK_VAL;
            __bf16* p = &pblk[pr * WHS_ROW + pc0];
            p[0] = (__bf16)(__expf(e0 - mv) * isv);
            p[1] = (__bf16)(__expf(e1 - mv) * isv);
            p[2] = (__bf16)(__expf(e2 - mv) * isv);
            p[3] = (__bf16)(__expf(e3 - mv) * isv);
        }
        __syncthreads();

        BF16x16 af[2];
#pragma unroll
        for (int mt = 0; mt < 2; mt++) {
            const int mm = 16 * mt + lr;
            af[mt].h[0] = *reinterpret_cast<const v8bf*>(&pblk[mm * WHS_ROW + 8 * lh]);
            af[mt].h[1] = *reinterpret_cast<const v8bf*>(&pblk[mm * WHS_ROW + 16 + 8 * lh]);
        }
#pragma unroll
        for (int nt = 0; nt < 4; nt++) {
            BF16x16 bf;
            const int nn = n0 + 16 * nt + lr;
            bf.h[0] = *reinterpret_cast<const v8bf*>(&whs[cur][nn * WHS_ROW + 8 * lh]);
            bf.h[1] = *reinterpret_cast<const v8bf*>(&whs[cur][nn * WHS_ROW + 16 + 8 * lh]);
#pragma unroll
            for (int mt = 0; mt < 2; mt++)
                acc[mt][nt].v = __builtin_amdgcn_wmma_f32_16x16x32_bf16(
                    false, af[mt].v, false, bf.v, (short)0, acc[mt][nt].v,
                    false, false);
        }
        __syncthreads();
    }

    // ELU epilogue + store
#pragma unroll
    for (int mt = 0; mt < 2; mt++)
#pragma unroll
        for (int nt = 0; nt < 4; nt++)
#pragma unroll
            for (int r = 0; r < 8; r++) {
                const int m_g = row0 + 16 * mt + r + 8 * lh;
                const int n_g = n0 + 16 * nt + lr;
                const float x = acc[mt][nt].e[r];
                out[(size_t)m_g * 512 + n_g] = (x > 0.f) ? x : (__expf(x) - 1.f);
            }
}

// ---------------------------------------------------------------------------
extern "C" void kernel_launch(void* const* d_in, const int* in_sizes, int n_in,
                              void* d_out, int out_size, void* d_ws, size_t ws_size,
                              hipStream_t stream) {
    const float* h   = (const float*)d_in[0];   // (8,2048,512) f32
    const int*   adj = (const int*)  d_in[1];   // (8,2048,2048) i32
    const float* W   = (const float*)d_in[2];   // (512,512) f32
    const float* a   = (const float*)d_in[3];   // (1024,1) f32
    float* out = (float*)d_out;                 // (8,2048,512) f32

    char* ws = (char*)d_ws;
    const size_t whbBytes = (size_t)16384 * 512 * 2;      // 16 MiB each
    __bf16* Whb  = (__bf16*)ws;                           // row-major
    __bf16* Whbt = (__bf16*)(ws + whbBytes);              // per-batch [n][i]
    float*  ei   = (float*)(ws + 2 * whbBytes);
    float*  ej   = ei + 16384;
    float*  rmax = ej + 16384;
    float*  rsinv= rmax + 16384;

    dim3 g1(16384 / 64, 512 / 64);
    k1_gemm  <<<g1,        256, 0, stream>>>(h, W, Whb, Whbt);
    k2_rowdots<<<16384 / 8, 256, 0, stream>>>(Whb, a, ei, ej);
    k3_stats <<<16384 / 8, 256, 0, stream>>>(adj, ei, ej, rmax, rsinv);
    k4_attn  <<<16384 / 32, 256, 0, stream>>>(adj, Whbt, ei, ej, rmax, rsinv, out);
}
